// SpectralConv3d_fast_25031069401539
// MI455X (gfx1250) — compile-verified, matrix operand order checked by reading, not run on hardware
//
#include <hip/hip_runtime.h>
#include <hip/hip_bf16.h>
#include <math.h>

// ---------------- problem constants ----------------
#define XDIM 64
#define YDIM 64
#define ZDIM 64
#define ZR   33               // rfft length along z
#define NM   (XDIM*YDIM*ZR)   // 135168 modes per channel
#define BATCH 4
#define CIN  8
#define COUT 8
#define NCH  (BATCH*CIN)      // 32 (b,c) planes

typedef __attribute__((ext_vector_type(16))) _Float16 v16h;
typedef __attribute__((ext_vector_type(8)))  float    v8f;

__device__ __forceinline__ v8f wmma_f16(v16h a, v16h b, v8f c) {
    // D = A(16x32 f16) * B(32x16 f16) + C(16x16 f32)
    return __builtin_amdgcn_wmma_f32_16x16x32_f16(
        /*neg_a=*/false, a, /*neg_b=*/false, b,
        /*c_mod=*/(short)0, c, /*reuse_a=*/false, /*reuse_b=*/false);
}

__device__ __forceinline__ float gelu_exact(float x) {
    return 0.5f * x * (1.0f + erff(x * 0.70710678118654752f));
}

// Wave-level LDS ordering fence: same-wave DS ops are executed in order by
// hardware; this only stops the compiler from reordering across the point.
__device__ __forceinline__ void wave_lds_sync() {
    asm volatile("" ::: "memory");
    __builtin_amdgcn_wave_barrier();
    asm volatile("" ::: "memory");
}

// Async-copy 16KB (4096 f32) from global to LDS using the gfx1250
// ASYNCcnt-tracked direct-to-LDS path (no VGPR round trip).
// 128 threads x 8 iters x 16B. Caller must __syncthreads() afterwards.
__device__ __forceinline__ void async_copy_16KB(const float* __restrict__ src,
                                                float* dstLDS, int tid) {
    const uint32_t ldsbase = (uint32_t)(uintptr_t)dstLDS; // flat->LDS: low 32 bits
    const uint64_t gbase   = (uint64_t)(uintptr_t)src;
#pragma unroll
    for (int it = 0; it < 8; ++it) {
        const uint32_t off = (uint32_t)((it * 128 + tid) * 16);
        const uint32_t lds = ldsbase + off;
        asm volatile("global_load_async_to_lds_b128 %0, %1, %2"
                     :: "v"(lds), "v"(off), "s"(gbase)
                     : "memory");
    }
    asm volatile("s_wait_asynccnt 0x0" ::: "memory");
}

// =====================================================================
// Kernel 1: per-mode MLP weight generation via WMMA.
//   grid feats (fx,fy,fz) -> 64 -> gelu -> 64 -> gelu -> 64 -> gelu -> 64
//   One wave = 16 modes (M tile). Hidden 64 = 4 N-tiles, K=64 = 2 WMMA chunks.
//   Weights are staged via async-to-LDS copies, then pre-swizzled into exact
//   WMMA B-fragment order so every fragment load is two ds_load_b128.
// =====================================================================
__global__ __launch_bounds__(128)
void mlp_weights_kernel(const float* __restrict__ win,  const float* __restrict__ bin,
                        const float* __restrict__ w1,   const float* __restrict__ b1,
                        const float* __restrict__ w2,   const float* __restrict__ b2,
                        const float* __restrict__ wout, const float* __restrict__ bout,
                        float* __restrict__ out)        // [NM][64]
{
    __shared__ __align__(16) float    stage[64 * 64];        // 16KB async staging
    __shared__ __align__(32) _Float16 sW0f[4 * 32 * 16];     // layer0 B frags (pad K->32)
    __shared__ __align__(32) _Float16 sWs[3][4 * 2 * 32 * 16]; // swizzled W1,W2,Wout
    __shared__ float    sB[4][64];
    __shared__ __align__(32) _Float16 sActA[4][16 * 64];
    __shared__ __align__(32) _Float16 sActB[4][16 * 64];

    const int tid  = threadIdx.x;
    const int wave = tid >> 5;
    const int lane = tid & 31;

    // biases
    for (int i = tid; i < 64; i += 128) {
        sB[0][i] = bin[i]; sB[1][i] = b1[i]; sB[2][i] = b2[i]; sB[3][i] = bout[i];
    }
    // layer-0 weight directly into zero-padded fragment layout (tiny: 3x64)
    for (int t = tid; t < 4 * 32 * 16; t += 128) {
        const int k  = t & 15;
        const int ln = (t >> 4) & 31;
        const int nt = (t >> 9) & 3;
        const int K   = ((ln >> 4) << 4) + k;            // 0..31
        const int col = (ln & 15) + nt * 16;             // 0..63
        sW0f[t] = (K < 3) ? (_Float16)win[K * 64 + col] : (_Float16)0.0f;
    }
    // big weights: async stage (global->LDS) then convert+swizzle to frag order
    const float* Wg[3] = { w1, w2, wout };
#pragma unroll 1
    for (int m_ = 0; m_ < 3; ++m_) {
        async_copy_16KB(Wg[m_], stage, tid);
        __syncthreads();   // all waves' async copies done (each waited its own cnt)
        for (int t = tid; t < 4 * 2 * 32 * 16; t += 128) {
            const int k  = t & 15;
            const int ln = (t >> 4) & 31;
            const int c  = (t >> 9) & 1;
            const int nt = (t >> 10) & 3;
            const int row = c * 32 + ((ln >> 4) << 4) + k;   // K index 0..63
            const int col = (ln & 15) + nt * 16;             // N index 0..63
            sWs[m_][t] = (_Float16)stage[row * 64 + col];
        }
        __syncthreads();   // staging reusable for next matrix
    }

    const int m0    = (blockIdx.x * 4 + wave) * 16;   // first mode of this wave
    const int M     = lane & 15;                      // A-matrix row owned by lane
    const int kbase = (lane < 16) ? 0 : 16;           // K sub-range per half-wave
    const int Mrow0 = (lane < 16) ? 0 : 8;            // D-matrix row base per half-wave
    const int ncol  = lane & 15;                      // D column within N tile

    // frequency features for this lane's mode
    const int mode = m0 + M;
    const int gx = mode / (YDIM * ZR);
    const int rr = mode % (YDIM * ZR);
    const int gy = rr / ZR;
    const int gz = rr % ZR;
    const float f0 = (float)(gx < 32 ? gx : gx - 64) * (1.0f / 64.0f);
    const float f1 = (float)(gy < 32 ? gy : gy - 64) * (1.0f / 64.0f);
    const float f2 = (float)gz * (1.0f / 64.0f);

    _Float16* cur = &sActA[wave][0];
    _Float16* nxt = &sActB[wave][0];

    // ---- layer 0: [16x3 (pad 32)] @ [3x64 (pad 32)] + gelu ----
    v16h a0;
#pragma unroll
    for (int k = 0; k < 16; ++k) {
        const int K = kbase + k;
        float v = (K == 0) ? f0 : (K == 1) ? f1 : (K == 2) ? f2 : 0.0f;
        a0[k] = (_Float16)v;
    }
#pragma unroll
    for (int nt = 0; nt < 4; ++nt) {
        const int col = ncol + nt * 16;
        const v16h b = *(const v16h*)&sW0f[(nt * 32 + lane) * 16];
        v8f c = {};
        c = wmma_f16(a0, b, c);
#pragma unroll
        for (int r = 0; r < 8; ++r) {
            float h = gelu_exact(c[r] + sB[0][col]);
            cur[(Mrow0 + r) * 64 + col] = (_Float16)h;
        }
    }
    wave_lds_sync();

    // ---- hidden layers 1 and 2: [16x64] @ [64x64] + gelu ----
#pragma unroll
    for (int layer = 0; layer < 2; ++layer) {
        const _Float16* W  = &sWs[layer][0];
        const float*    bs = &sB[layer + 1][0];
        const v16h aLo = *(const v16h*)&cur[M * 64 + kbase];
        const v16h aHi = *(const v16h*)&cur[M * 64 + 32 + kbase];
#pragma unroll
        for (int nt = 0; nt < 4; ++nt) {
            const int col = ncol + nt * 16;
            const v16h bLo = *(const v16h*)&W[((nt * 2 + 0) * 32 + lane) * 16];
            const v16h bHi = *(const v16h*)&W[((nt * 2 + 1) * 32 + lane) * 16];
            v8f c = {};
            c = wmma_f16(aLo, bLo, c);
            c = wmma_f16(aHi, bHi, c);
#pragma unroll
            for (int r = 0; r < 8; ++r) {
                float h = gelu_exact(c[r] + bs[col]);
                nxt[(Mrow0 + r) * 64 + col] = (_Float16)h;
            }
        }
        _Float16* t = cur; cur = nxt; nxt = t;
        wave_lds_sync();
    }

    // ---- output layer: [16x64] @ [64x64] + bias, store f32 ----
    {
        const v16h aLo = *(const v16h*)&cur[M * 64 + kbase];
        const v16h aHi = *(const v16h*)&cur[M * 64 + 32 + kbase];
#pragma unroll
        for (int nt = 0; nt < 4; ++nt) {
            const int col = ncol + nt * 16;
            const v16h bLo = *(const v16h*)&sWs[2][((nt * 2 + 0) * 32 + lane) * 16];
            const v16h bHi = *(const v16h*)&sWs[2][((nt * 2 + 1) * 32 + lane) * 16];
            v8f c = {};
            c = wmma_f16(aLo, bLo, c);
            c = wmma_f16(aHi, bHi, c);
#pragma unroll
            for (int r = 0; r < 8; ++r) {
                const int Mr = Mrow0 + r;
                out[(size_t)(m0 + Mr) * 64 + col] = c[r] + sB[3][col];
            }
        }
    }
}

// =====================================================================
// 64-point radix-2 DIT FFT in LDS; one wave (32 lanes) per private line.
// Same-wave LDS ops are in-order, so only compiler-level fences needed.
// sign = -1 forward, +1 inverse (unnormalized).
// =====================================================================
__device__ __forceinline__ void fft64(float2* s, int lane, float sign) {
    // bit-reversal permute: new[i] = old[rev6(i)]
    const int i0 = 2 * lane, i1 = 2 * lane + 1;
    const int r0 = (int)(__brev((unsigned)i0) >> 26);
    const int r1 = (int)(__brev((unsigned)i1) >> 26);
    float2 a0 = s[r0];
    float2 a1 = s[r1];
    wave_lds_sync();
    s[i0] = a0; s[i1] = a1;
    wave_lds_sync();
#pragma unroll
    for (int st = 1; st <= 6; ++st) {
        const int m    = 1 << st;
        const int half = m >> 1;
        const int j    = lane & (half - 1);
        const int k    = ((lane & ~(half - 1)) << 1) + j;
        float sn, cs;
        __sincosf(sign * 6.28318530717958647692f * (float)j / (float)m, &sn, &cs);
        float2 u = s[k];
        float2 t = s[k + half];
        float2 tw = make_float2(t.x * cs - t.y * sn, t.x * sn + t.y * cs);
        s[k]        = make_float2(u.x + tw.x, u.y + tw.y);
        s[k + half] = make_float2(u.x - tw.x, u.y - tw.y);
        wave_lds_sync();
    }
}

// ---- pass 1: real input, FFT along z, keep 33 bins ----
__global__ __launch_bounds__(256)
void fft_z_r2c_kernel(const float* __restrict__ x, float2* __restrict__ spec) {
    __shared__ float2 sh[8][64];
    const int wave = threadIdx.x >> 5;
    const int lane = threadIdx.x & 31;
    const int L = blockIdx.x * 8 + wave;          // 0 .. 131071
    const int bc = L >> 12;
    const int gx = (L >> 6) & 63;
    const int gy = L & 63;
    const float* src = x + ((((size_t)bc * 64 + gx) * 64 + gy) * 64);
    sh[wave][2 * lane]     = make_float2(src[2 * lane], 0.0f);
    sh[wave][2 * lane + 1] = make_float2(src[2 * lane + 1], 0.0f);
    wave_lds_sync();
    fft64(&sh[wave][0], lane, -1.0f);
    float2* dst = spec + (size_t)bc * NM + ((size_t)(gx * 64 + gy)) * ZR;
    dst[lane] = sh[wave][lane];
    if (lane == 0) dst[32] = sh[wave][32];
}

// ---- passes along y (axis=1) or x (axis=0), forward or inverse ----
__global__ __launch_bounds__(256)
void fft_axis_kernel(float2* __restrict__ data, float sign, int axis) {
    __shared__ float2 sh[8][64];
    const int wave = threadIdx.x >> 5;
    const int lane = threadIdx.x & 31;
    const int L  = blockIdx.x * 8 + wave;         // 0 .. 67583
    const int bc = L / (64 * ZR);
    const int rem = L % (64 * ZR);
    size_t off;
    int stride;
    if (axis == 1) { // along y: fixed (x,z); element index (x*64+y)*33+z
        const int gx = rem / ZR, gz = rem % ZR;
        off = (size_t)bc * NM + (size_t)gx * (64 * ZR) + gz;
        stride = ZR;
    } else {         // along x: fixed (y,z)
        off = (size_t)bc * NM + (size_t)rem;
        stride = 64 * ZR;
    }
    float2* base = data + off;
    sh[wave][2 * lane]     = base[(size_t)(2 * lane) * stride];
    sh[wave][2 * lane + 1] = base[(size_t)(2 * lane + 1) * stride];
    wave_lds_sync();
    fft64(&sh[wave][0], lane, sign);
    base[(size_t)(2 * lane) * stride]     = sh[wave][2 * lane];
    base[(size_t)(2 * lane + 1) * stride] = sh[wave][2 * lane + 1];
}

// ---- per-mode 8x8 complex channel contraction ----
__global__ __launch_bounds__(256)
void mix_kernel(const float2* __restrict__ spec, const float* __restrict__ wRe,
                const float* __restrict__ wIm, float2* __restrict__ ospec) {
    const size_t t = (size_t)blockIdx.x * 256 + threadIdx.x;  // < 32*NM
    const int m  = (int)(t % NM);
    const int bo = (int)(t / NM);
    const int b = bo >> 3, o = bo & 7;
    const float* wr = wRe + (size_t)m * 64 + o * CIN;
    const float* wi = wIm + (size_t)m * 64 + o * CIN;
    float sr = 0.0f, si = 0.0f;
#pragma unroll
    for (int i = 0; i < CIN; ++i) {
        float2 a = spec[(size_t)(b * CIN + i) * NM + m];
        const float xr = wr[i], xi = wi[i];
        sr += a.x * xr - a.y * xi;
        si += a.x * xi + a.y * xr;
    }
    ospec[(size_t)(b * COUT + o) * NM + m] = make_float2(sr, si);
}

// ---- final pass: Hermitian-extend along z, inverse FFT, real out + scale ----
__global__ __launch_bounds__(256)
void irfft_z_kernel(const float2* __restrict__ ospec, float* __restrict__ out) {
    __shared__ float2 sh[8][64];
    const int wave = threadIdx.x >> 5;
    const int lane = threadIdx.x & 31;
    const int L = blockIdx.x * 8 + wave;          // 0 .. 131071
    const int bc = L >> 12;
    const int gx = (L >> 6) & 63;
    const int gy = L & 63;
    const float2* src = ospec + (size_t)bc * NM + ((size_t)(gx * 64 + gy)) * ZR;
    float2 v = src[lane];
    sh[wave][lane] = v;
    if (lane > 0) sh[wave][64 - lane] = make_float2(v.x, -v.y);
    if (lane == 0) sh[wave][32] = src[32];
    wave_lds_sync();
    fft64(&sh[wave][0], lane, +1.0f);
    const float scale = 1.0f / (float)(XDIM * YDIM * ZDIM);
    float* dst = out + ((((size_t)bc * 64 + gx) * 64 + gy) * 64);
    dst[2 * lane]     = sh[wave][2 * lane].x * scale;
    dst[2 * lane + 1] = sh[wave][2 * lane + 1].x * scale;
}

// =====================================================================
extern "C" void kernel_launch(void* const* d_in, const int* in_sizes, int n_in,
                              void* d_out, int out_size, void* d_ws, size_t ws_size,
                              hipStream_t stream) {
    const float* x      = (const float*)d_in[0];
    const float* wr_in  = (const float*)d_in[1];
    const float* br_in  = (const float*)d_in[2];
    const float* wr_h1  = (const float*)d_in[3];
    const float* br_h1  = (const float*)d_in[4];
    const float* wr_h2  = (const float*)d_in[5];
    const float* br_h2  = (const float*)d_in[6];
    const float* wr_out = (const float*)d_in[7];
    const float* br_out = (const float*)d_in[8];
    const float* wi_in  = (const float*)d_in[9];
    const float* bi_in  = (const float*)d_in[10];
    const float* wi_h1  = (const float*)d_in[11];
    const float* bi_h1  = (const float*)d_in[12];
    const float* wi_h2  = (const float*)d_in[13];
    const float* bi_h2  = (const float*)d_in[14];
    const float* wi_out = (const float*)d_in[15];
    const float* bi_out = (const float*)d_in[16];
    float* out = (float*)d_out;

    // workspace partition
    float*  wRe   = (float*)d_ws;                       // NM*64 f32
    float*  wIm   = wRe + (size_t)NM * 64;              // NM*64 f32
    float2* spec  = (float2*)(wIm + (size_t)NM * 64);   // 32*NM c64 (x_ft)
    float2* ospec = spec + (size_t)NCH * NM;            // 32*NM c64 (out_ft)

    // 1) spectral weights via WMMA MLPs (re and im)
    mlp_weights_kernel<<<NM / 64, 128, 0, stream>>>(
        wr_in, br_in, wr_h1, br_h1, wr_h2, br_h2, wr_out, br_out, wRe);
    mlp_weights_kernel<<<NM / 64, 128, 0, stream>>>(
        wi_in, bi_in, wi_h1, bi_h1, wi_h2, bi_h2, wi_out, bi_out, wIm);

    // 2) forward rfftn: z (r2c), then y, then x
    fft_z_r2c_kernel<<<(NCH * 64 * 64) / 8, 256, 0, stream>>>(x, spec);
    fft_axis_kernel<<<(NCH * 64 * ZR) / 8, 256, 0, stream>>>(spec, -1.0f, 1);
    fft_axis_kernel<<<(NCH * 64 * ZR) / 8, 256, 0, stream>>>(spec, -1.0f, 0);

    // 3) per-mode channel mixing
    mix_kernel<<<(NCH * NM) / 256, 256, 0, stream>>>(spec, wRe, wIm, ospec);

    // 4) inverse: x, y, then irfft z (with 1/N^3 scale)
    fft_axis_kernel<<<(NCH * 64 * ZR) / 8, 256, 0, stream>>>(ospec, +1.0f, 0);
    fft_axis_kernel<<<(NCH * 64 * ZR) / 8, 256, 0, stream>>>(ospec, +1.0f, 1);
    irfft_z_kernel<<<(NCH * 64 * 64) / 8, 256, 0, stream>>>(ospec, out);

    (void)in_sizes; (void)n_in; (void)out_size; (void)ws_size;
}
